// GCN_26431228740293
// MI455X (gfx1250) — matrix operand order repaired and checked by reading, not compiled, not run on hardware
//
#include <hip/hip_runtime.h>

#define N_NODES 100000
#define N_EDGES 1600000
#define F_H 128
#define F_O 64
#define MT 2              // M-tiles (of 16 rows) per GEMM block -> 32 rows/block

typedef __bf16 bf16;
typedef __attribute__((ext_vector_type(16))) __bf16 v16bf;
typedef __attribute__((ext_vector_type(8)))  float  v8f;

// low 32 bits of a flat LDS pointer are the LDS byte offset (ISA: LDS aperture
// truncates addr[31:0]); usable directly as the async-copy LDS destination.
__device__ __forceinline__ unsigned lds_off(const void* p) {
    return (unsigned)(uintptr_t)p;
}

__device__ __forceinline__ void async_ld_b128(unsigned lds, const void* gaddr) {
    asm volatile("global_load_async_to_lds_b128 %0, %1, off"
                 :: "v"(lds), "v"(gaddr) : "memory");
}

__device__ __forceinline__ void wait_async() {
    asm volatile("s_wait_asynccnt 0" ::: "memory");
}

// ---------------- degree / normalization ----------------

__global__ void init_deg_kernel(float* deg) {
    int i = blockIdx.x * blockDim.x + threadIdx.x;
    if (i < N_NODES) deg[i] = 1.0f;  // self-loop contribution
}

__global__ void deg_atomic_kernel(const long long* __restrict__ dst, float* deg) {
    int e = blockIdx.x * blockDim.x + threadIdx.x;
    if (e < N_EDGES) atomicAdd(&deg[(int)dst[e]], 1.0f);
}

__global__ void finalize_dinv_kernel(float* deg) {
    int i = blockIdx.x * blockDim.x + threadIdx.x;
    if (i < N_NODES) deg[i] = rsqrtf(deg[i]);  // deg >= 1 always
}

// ---------------- precision prep: f32 -> bf16 ----------------

__global__ void cvt_bf16_kernel(const float* __restrict__ in, bf16* __restrict__ out, int n) {
    int i = blockIdx.x * blockDim.x + threadIdx.x;
    if (i < n) out[i] = (bf16)in[i];
}

// W[k][n] (row-major [128][NOUT]) -> WT[n][k] bf16, so GEMM B-columns are contiguous
__global__ void transpose_bf16_kernel(const float* __restrict__ W, bf16* __restrict__ WT, int nout) {
    int i = blockIdx.x * blockDim.x + threadIdx.x;
    if (i < 128 * nout) {
        int k = i / nout, n = i % nout;
        WT[n * 128 + k] = (bf16)W[i];
    }
}

// ---------------- bf16 WMMA GEMM: Y[M,NOUT] = X[M,128] @ W[128,NOUT] ----------------
// X, WT already bf16 in global. One block (4 waves) per 32-row M-block; A-tile and
// WT staged to LDS with CDNA5 async copies (ASYNCcnt), fragments in contiguous k order
// (k = 16*half + j) for A (16x32) and B (32x16), K=128 swept in 4 WMMA ops per tile.

template <int NOUT>
__global__ __launch_bounds__(128) void gemm_bf16_wmma(const bf16* __restrict__ X,
                                                      const bf16* __restrict__ WT,
                                                      float* __restrict__ Y) {
    __shared__ __align__(16) bf16 As[MT * 16 * 128];   // 8 KB  : As[m*128 + k]
    __shared__ __align__(16) bf16 WsT[NOUT * 128];     // 32/16 KB : WsT[n*128 + k]

    const int tid = threadIdx.x;
    const int m0  = blockIdx.x * (MT * 16);

    // async-stage A tile (contiguous MT*16 rows) and transposed W, 16B per lane-op
    {
        const unsigned aBase = lds_off(As);
        const bf16*    ag    = X + (size_t)m0 * 128;
#pragma unroll
        for (int c = tid; c < (MT * 16 * 128) / 8; c += 128)
            async_ld_b128(aBase + c * 16, ag + c * 8);
        const unsigned wBase = lds_off(WsT);
#pragma unroll
        for (int c = tid; c < (NOUT * 128) / 8; c += 128)
            async_ld_b128(wBase + c * 16, WT + c * 8);
        wait_async();
        __syncthreads();
    }

    const int wave  = tid >> 5;
    const int lane  = tid & 31;
    const int half  = lane >> 4;   // 0: lanes 0-15, 1: lanes 16-31
    const int l16   = lane & 15;
    const int koffA = half * 16;   // A: lanes 0-15 hold K 0-15; 16-31 hold K 16-31
    const int koffB = half * 16;   // B: lanes 0-15 hold K 0-15; 16-31 hold K 16-31

    constexpr int NT = NOUT / 16;
    for (int t = wave; t < MT * NT; t += 4) {
        const int mt = t / NT;
        const int n0 = (t % NT) * 16;
        v8f acc = {};
#pragma unroll
        for (int kc = 0; kc < 4; ++kc) {   // K = 128 in chunks of 32
            v16bf a, b;
            const bf16* ap = &As[(mt * 16 + l16) * 128 + kc * 32 + koffA];
#pragma unroll
            for (int j = 0; j < 16; ++j) a[j] = ap[j];
            const bf16* bp = &WsT[(n0 + l16) * 128 + kc * 32 + koffB];
#pragma unroll
            for (int j = 0; j < 16; ++j) b[j] = bp[j];
            acc = __builtin_amdgcn_wmma_f32_16x16x32_bf16(
                false, a, false, b, (short)0, acc, false, false);
        }
        // C/D layout: acc[r] -> (M = r + 8*half, N = lane&15)
#pragma unroll
        for (int r = 0; r < 8; ++r) {
            Y[(size_t)(m0 + mt * 16 + r + half * 8) * NOUT + n0 + l16] = acc[r];
        }
    }
}

// ---------------- edge-parallel aggregation (one wave per edge) ----------------

template <int F>
__global__ __launch_bounds__(256) void agg_atomic_kernel(const long long* __restrict__ srcA,
                                                         const long long* __restrict__ dstA,
                                                         const float* __restrict__ dinv,
                                                         const float* __restrict__ feat,
                                                         float* __restrict__ out) {
    const int lane = threadIdx.x & 31;
    const int wid  = blockIdx.x * (blockDim.x >> 5) + (threadIdx.x >> 5);
    if (wid >= N_EDGES) return;
    const int s = (int)srcA[wid];
    const int d = (int)dstA[wid];
    const float coef = dinv[s] * dinv[d];
    const float* fp = feat + (size_t)s * F;
    float*       op = out  + (size_t)d * F;
#pragma unroll
    for (int j = lane; j < F; j += 32) {
        atomicAdd(&op[j], fp[j] * coef);   // no-return f32 atomic (STOREcnt)
    }
}

// ---------------- self-loop + bias (+ReLU) epilogues ----------------

// layer 1: h = relu(acc + xw*dinv^2 + b); written directly as bf16 for GEMM2
__global__ void selfloop_relu_kernel(const float* __restrict__ xw,
                                     const float* __restrict__ acc,
                                     const float* __restrict__ dinv,
                                     const float* __restrict__ b,
                                     bf16* __restrict__ hbf) {
    int i = blockIdx.x * blockDim.x + threadIdx.x;
    if (i >= N_NODES * F_H) return;
    int row = i >> 7, col = i & 127;
    float di = dinv[row];
    float v = acc[i] + xw[i] * di * di + b[col];
    hbf[i] = (bf16)fmaxf(v, 0.0f);
}

// layer 2: out += hw*dinv^2 + b (f32, no relu)
__global__ void selfloop_out_kernel(const float* __restrict__ hw,
                                    const float* __restrict__ dinv,
                                    const float* __restrict__ b,
                                    float* __restrict__ out) {
    int i = blockIdx.x * blockDim.x + threadIdx.x;
    if (i >= N_NODES * F_O) return;
    int row = i >> 6, col = i & 63;
    float di = dinv[row];
    out[i] = out[i] + hw[i] * di * di + b[col];
}

// ---------------- launch ----------------

extern "C" void kernel_launch(void* const* d_in, const int* in_sizes, int n_in,
                              void* d_out, int out_size, void* d_ws, size_t ws_size,
                              hipStream_t stream) {
    const float*     x   = (const float*)d_in[0];
    const long long* ei  = (const long long*)d_in[1];
    const float*     W1  = (const float*)d_in[2];
    const float*     b1  = (const float*)d_in[3];
    const float*     W2  = (const float*)d_in[4];
    const float*     b2  = (const float*)d_in[5];
    float*           out = (float*)d_out;

    const long long* srcA = ei;
    const long long* dstA = ei + N_EDGES;

    // workspace layout: dinv f32 | bufA f32 (xw->hw) | bufB f32 (h_acc) |
    //                   xbf bf16 (x_bf16, reused as h_bf16) | WT1 bf16 | WT2 bf16
    float* dinv = (float*)d_ws;
    float* bufA = dinv + ((N_NODES + 255) & ~255);          // 16B-aligned offsets
    float* bufB = bufA + (size_t)N_NODES * 128;
    bf16*  xbf  = (bf16*)(bufB + (size_t)N_NODES * 128);
    bf16*  WT1  = xbf + (size_t)N_NODES * 128;
    bf16*  WT2  = WT1 + 128 * F_H;

    hipMemsetAsync(bufB, 0, (size_t)N_NODES * F_H * sizeof(float), stream);
    hipMemsetAsync(out,  0, (size_t)N_NODES * F_O * sizeof(float), stream);

    // degree + symmetric normalization
    init_deg_kernel<<<(N_NODES + 255) / 256, 256, 0, stream>>>(dinv);
    deg_atomic_kernel<<<(N_EDGES + 255) / 256, 256, 0, stream>>>(dstA, dinv);
    finalize_dinv_kernel<<<(N_NODES + 255) / 256, 256, 0, stream>>>(dinv);

    // precision prep
    cvt_bf16_kernel<<<(N_NODES * F_H + 255) / 256, 256, 0, stream>>>(x, xbf, N_NODES * F_H);
    transpose_bf16_kernel<<<(128 * F_H + 255) / 256, 256, 0, stream>>>(W1, WT1, F_H);
    transpose_bf16_kernel<<<(128 * F_O + 255) / 256, 256, 0, stream>>>(W2, WT2, F_O);

    // layer 1: xw = x @ W1 ; aggregate ; self-loop + bias + relu (-> bf16 h)
    gemm_bf16_wmma<128><<<N_NODES / (MT * 16), 128, 0, stream>>>(xbf, WT1, bufA);
    agg_atomic_kernel<128><<<N_EDGES / 8, 256, 0, stream>>>(srcA, dstA, dinv, bufA, bufB);
    selfloop_relu_kernel<<<(N_NODES * F_H) / 256, 256, 0, stream>>>(bufA, bufB, dinv, b1, xbf);

    // layer 2: hw = h @ W2 ; aggregate into d_out ; self-loop + bias
    gemm_bf16_wmma<64><<<N_NODES / (MT * 16), 128, 0, stream>>>(xbf, WT2, bufA);
    agg_atomic_kernel<64><<<N_EDGES / 8, 256, 0, stream>>>(srcA, dstA, dinv, bufA, out);
    selfloop_out_kernel<<<(N_NODES * F_O) / 256, 256, 0, stream>>>(bufA, dinv, b2, out);
}
